// SABlock_88450556494563
// MI455X (gfx1250) — matrix operand order repaired and checked by reading, compiled, not run on hardware
//
#include <hip/hip_runtime.h>

typedef __attribute__((ext_vector_type(16))) _Float16 v16h;
typedef __attribute__((ext_vector_type(8)))  _Float16 v8h;
typedef __attribute__((ext_vector_type(8)))  float    v8f;
typedef int v4i_vs __attribute__((vector_size(16)));   // pointee type the async builtin wants

#define SEQ   2048
#define NH    12
#define HD    64
#define HID   768
#define BATCH 2
#define ROWS  (BATCH * SEQ)   // 4096

#if __has_builtin(__builtin_amdgcn_global_load_async_to_lds_b128) && \
    __has_builtin(__builtin_amdgcn_s_wait_asynccnt)
#define USE_ASYNC_LDS 1
#else
#define USE_ASYNC_LDS 0
#endif

__device__ __forceinline__ v8f wmma16(v16h a, v16h b, v8f c) {
  return __builtin_amdgcn_wmma_f32_16x16x32_f16(
      /*neg_a=*/false, a, /*neg_b=*/false, b,
      /*c_mod=*/(short)0, c, /*reuse_a=*/false, /*reuse_b=*/false);
}

// A fragment: 16(M) x 32(K), row-major source, leading dim lda (halves).
// Lanes 0-15 -> M=lane, K halves {0..7,16..23}; lanes 16-31 -> {8..15,24..31}.
__device__ __forceinline__ v16h load_a(const _Float16* A, int lda, int lane) {
  int row = lane & 15;
  int ko  = (lane >> 4) * 8;
  const _Float16* p = A + row * lda + ko;
  union { v16h v; v8h h[2]; } u;
  u.h[0] = *(const v8h*)(p);
  u.h[1] = *(const v8h*)(p + 16);
  return u.v;
}

// B fragment: 32(K) x 16(N) from Bt stored N-major (Bt[n][k]), leading dim ldb.
// Lanes 0-15: N=lane, K {0..7,16..23}; lanes 16-31: N=lane-16, K {8..15,24..31}.
__device__ __forceinline__ v16h load_bt(const _Float16* Bt, int ldb, int lane) {
  int n  = lane & 15;
  int ko = (lane >> 4) * 8;
  const _Float16* p = Bt + n * ldb + ko;
  union { v16h v; v8h h[2]; } u;
  u.h[0] = *(const v8h*)(p);
  u.h[1] = *(const v8h*)(p + 16);
  return u.v;
}

// 16-byte global -> LDS copy; async (ASYNCcnt) when available.
__device__ __forceinline__ void cp16(const _Float16* g, _Float16* l) {
#if USE_ASYNC_LDS
  v4i_vs* gp = (v4i_vs*)g;   // generic reinterpret (drops const)
  v4i_vs* lp = (v4i_vs*)l;
  __builtin_amdgcn_global_load_async_to_lds_b128(
      (__attribute__((address_space(1))) v4i_vs*)gp,
      (__attribute__((address_space(3))) v4i_vs*)lp, 0, 0);
#else
  *(v8h*)l = *(const v8h*)g;
#endif
}

__device__ __forceinline__ void wait_async_lds() {
#if USE_ASYNC_LDS
  __builtin_amdgcn_s_wait_asynccnt(0);
#endif
}

__global__ void cast_f32_f16(const float* __restrict__ in, _Float16* __restrict__ out, int n) {
  int i = blockIdx.x * blockDim.x + threadIdx.x;
  int stride = gridDim.x * blockDim.x;
  for (; i < n; i += stride) out[i] = (_Float16)in[i];
}

// QKV GEMM: [4096 x 768] @ [768 x 2304] (Bt = w_qkv row-major). Wave tile 64x64.
__global__ void __launch_bounds__(128)
qkv_gemm(const _Float16* __restrict__ xh, const _Float16* __restrict__ wh,
         _Float16* __restrict__ qh, _Float16* __restrict__ kh, _Float16* __restrict__ vth) {
  const int lane = threadIdx.x & 31;
  const int widx = blockIdx.x * 4 + (threadIdx.x >> 5);
  const int tiles_n = (3 * HID) / 64;          // 36
  const int tm = widx / tiles_n, tn = widx % tiles_n;
  const int m0 = tm * 64, n0 = tn * 64;

  v8f acc[4][4] = {};
  for (int k0 = 0; k0 < HID; k0 += 32) {
    v16h a[4];
#pragma unroll
    for (int i = 0; i < 4; ++i)
      a[i] = load_a(xh + (size_t)(m0 + i * 16) * HID + k0, HID, lane);
#pragma unroll
    for (int j = 0; j < 4; ++j) {
      v16h b = load_bt(wh + (size_t)(n0 + j * 16) * HID + k0, HID, lane);
#pragma unroll
      for (int i = 0; i < 4; ++i)
        acc[i][j] = wmma16(a[i], b, acc[i][j]);
    }
  }
#pragma unroll
  for (int i = 0; i < 4; ++i)
#pragma unroll
    for (int j = 0; j < 4; ++j)
#pragma unroll
      for (int r = 0; r < 8; ++r) {
        int gm = m0 + i * 16 + ((lane >> 4) << 3) + r;
        int gn = n0 + j * 16 + (lane & 15);
        int b  = gm >> 11, ns = gm & (SEQ - 1);
        int sel = gn / HID, rem = gn % HID;
        int h = rem >> 6, d = rem & 63;
        int bh = b * NH + h;
        float v = acc[i][j][r];
        if (sel == 0)      qh[(((size_t)bh * SEQ + ns) << 6) + d] = (_Float16)(v * 0.125f); // pre-scale Q
        else if (sel == 1) kh[(((size_t)bh * SEQ + ns) << 6) + d] = (_Float16)v;
        else               vth[((size_t)bh * HD + d) * SEQ + ns]  = (_Float16)v;            // V transposed
      }
}

// Stage one 32-step K/V tile for a whole block into LDS (K: 32x64 contiguous 4KB;
// V^T: 64 rows x 32 cols, row stride SEQ). 4 x b128 per thread.
__device__ __forceinline__ void stage_kv(const _Float16* __restrict__ Kg,
                                         const _Float16* __restrict__ Vtg,
                                         int j0, _Float16* lds, int tid) {
  const _Float16* ksrc = Kg + (size_t)j0 * HD;
#pragma unroll
  for (int r = 0; r < 2; ++r) {
    int c = tid + r * 128;                       // 16B chunk id, 0..255
    cp16(ksrc + c * 8, lds + c * 8);
  }
#pragma unroll
  for (int r = 0; r < 2; ++r) {
    int c = tid + r * 128;
    int row = c >> 2, sub = c & 3;
    cp16(Vtg + (size_t)row * SEQ + j0 + sub * 8, lds + 2048 + row * 32 + sub * 8);
  }
}

// Flash attention: block = one (b,h), 4 waves x 32 query rows; K/V tiles shared via
// async LDS staging, double buffered.
__global__ void __launch_bounds__(128)
attn(const _Float16* __restrict__ qh, const _Float16* __restrict__ kh,
     const _Float16* __restrict__ vth, _Float16* __restrict__ yh) {
  __shared__ __align__(32) _Float16 kvbuf[2][4096];   // per buffer: K tile (2048) + V tile (2048)
  __shared__ __align__(32) _Float16 pbuf[4][32 * 32]; // per-wave P staging
  const int tid  = threadIdx.x;
  const int lane = tid & 31;
  const int warp = tid >> 5;
  const int bh = blockIdx.x >> 4;               // 16 row-groups per (b,h)
  const int m0 = ((blockIdx.x & 15) * 4 + warp) * 32;
  const int b = bh / NH, h = bh % NH;

  const _Float16* Q  = qh  + (size_t)bh * SEQ * HD;
  const _Float16* K  = kh  + (size_t)bh * SEQ * HD;
  const _Float16* Vt = vth + (size_t)bh * HD * SEQ;
  _Float16* pl = pbuf[warp];

  v16h aq[2][2];
#pragma unroll
  for (int i = 0; i < 2; ++i)
#pragma unroll
    for (int kd = 0; kd < 2; ++kd)
      aq[i][kd] = load_a(Q + (size_t)(m0 + i * 16) * HD + kd * 32, HD, lane);

  v8f o[2][4] = {};
  float mrow[2][8], lrow[2][8];
#pragma unroll
  for (int i = 0; i < 2; ++i)
#pragma unroll
    for (int r = 0; r < 8; ++r) { mrow[i][r] = -3.0e38f; lrow[i][r] = 0.f; }

  stage_kv(K, Vt, 0, kvbuf[0], tid);
  int cur = 0;

  for (int j0 = 0; j0 < SEQ; j0 += 32) {
    wait_async_lds();
    __syncthreads();                 // buffer[cur] ready; buffer[cur^1] free to overwrite
    if (j0 + 32 < SEQ) stage_kv(K, Vt, j0 + 32, kvbuf[cur ^ 1], tid);
    const _Float16* ldsK = kvbuf[cur];
    const _Float16* ldsV = kvbuf[cur] + 2048;

    v8f s[2][2] = {};
#pragma unroll
    for (int tj = 0; tj < 2; ++tj)
#pragma unroll
      for (int kd = 0; kd < 2; ++kd) {
        v16h bk = load_bt(ldsK + (tj * 16) * HD + kd * 32, HD, lane);
        s[0][tj] = wmma16(aq[0][kd], bk, s[0][tj]);
        s[1][tj] = wmma16(aq[1][kd], bk, s[1][tj]);
      }

    float scl[2][8];
#pragma unroll
    for (int i = 0; i < 2; ++i)
#pragma unroll
      for (int r = 0; r < 8; ++r) {
        float v = fmaxf(s[i][0][r], s[i][1][r]);
        v = fmaxf(v, __shfl_xor(v, 1));
        v = fmaxf(v, __shfl_xor(v, 2));
        v = fmaxf(v, __shfl_xor(v, 4));
        v = fmaxf(v, __shfl_xor(v, 8));
        float mn = fmaxf(mrow[i][r], v);
        scl[i][r] = __expf(mrow[i][r] - mn);
        mrow[i][r] = mn;
      }
#pragma unroll
    for (int i = 0; i < 2; ++i)
#pragma unroll
      for (int tj = 0; tj < 2; ++tj)
#pragma unroll
        for (int r = 0; r < 8; ++r)
          s[i][tj][r] = __expf(s[i][tj][r] - mrow[i][r]);
#pragma unroll
    for (int i = 0; i < 2; ++i)
#pragma unroll
      for (int r = 0; r < 8; ++r) {
        float v = s[i][0][r] + s[i][1][r];
        v += __shfl_xor(v, 1);
        v += __shfl_xor(v, 2);
        v += __shfl_xor(v, 4);
        v += __shfl_xor(v, 8);
        lrow[i][r] = lrow[i][r] * scl[i][r] + v;
      }
#pragma unroll
    for (int i = 0; i < 2; ++i)
#pragma unroll
      for (int dt = 0; dt < 4; ++dt)
#pragma unroll
        for (int r = 0; r < 8; ++r)
          o[i][dt][r] *= scl[i][r];

    // C-layout P -> per-wave LDS -> A-layout fragments (same-wave DS ops are in-order)
#pragma unroll
    for (int i = 0; i < 2; ++i)
#pragma unroll
      for (int tj = 0; tj < 2; ++tj)
#pragma unroll
        for (int r = 0; r < 8; ++r) {
          int row = i * 16 + ((lane >> 4) << 3) + r;
          int col = tj * 16 + (lane & 15);
          pl[row * 32 + col] = (_Float16)s[i][tj][r];
        }

    v16h bv[4];
#pragma unroll
    for (int dt = 0; dt < 4; ++dt)
      bv[dt] = load_bt(ldsV + dt * 16 * 32, 32, lane);
#pragma unroll
    for (int i = 0; i < 2; ++i) {
      v16h ap = load_a(pl + i * 16 * 32, 32, lane);
#pragma unroll
      for (int dt = 0; dt < 4; ++dt)
        o[i][dt] = wmma16(ap, bv[dt], o[i][dt]);
    }
    cur ^= 1;
  }

#pragma unroll
  for (int i = 0; i < 2; ++i)
#pragma unroll
    for (int r = 0; r < 8; ++r) {
      float inv = 1.0f / lrow[i][r];
      int row = m0 + i * 16 + ((lane >> 4) << 3) + r;
#pragma unroll
      for (int dt = 0; dt < 4; ++dt) {
        int col = dt * 16 + (lane & 15);
        yh[((size_t)(b * SEQ + row)) * HID + h * HD + col] = (_Float16)(o[i][dt][r] * inv);
      }
    }
}

// Output projection: [4096 x 768] @ [768 x 768] (Bt = w_out row-major) + bias. Wave tile 64x64.
__global__ void __launch_bounds__(128)
out_proj(const _Float16* __restrict__ yh, const _Float16* __restrict__ wh,
         const float* __restrict__ bias, float* __restrict__ out) {
  const int lane = threadIdx.x & 31;
  const int widx = blockIdx.x * 4 + (threadIdx.x >> 5);
  const int tiles_n = HID / 64;                // 12
  const int tm = widx / tiles_n, tn = widx % tiles_n;
  const int m0 = tm * 64, n0 = tn * 64;

  v8f acc[4][4] = {};
  for (int k0 = 0; k0 < HID; k0 += 32) {
    v16h a[4];
#pragma unroll
    for (int i = 0; i < 4; ++i)
      a[i] = load_a(yh + (size_t)(m0 + i * 16) * HID + k0, HID, lane);
#pragma unroll
    for (int j = 0; j < 4; ++j) {
      v16h bfr = load_bt(wh + (size_t)(n0 + j * 16) * HID + k0, HID, lane);
#pragma unroll
      for (int i = 0; i < 4; ++i)
        acc[i][j] = wmma16(a[i], bfr, acc[i][j]);
    }
  }
#pragma unroll
  for (int i = 0; i < 4; ++i)
#pragma unroll
    for (int j = 0; j < 4; ++j)
#pragma unroll
      for (int r = 0; r < 8; ++r) {
        int gm = m0 + i * 16 + ((lane >> 4) << 3) + r;
        int gn = n0 + j * 16 + (lane & 15);
        out[(size_t)gm * HID + gn] = acc[i][j][r] + bias[gn];
      }
}

extern "C" void kernel_launch(void* const* d_in, const int* in_sizes, int n_in,
                              void* d_out, int out_size, void* d_ws, size_t ws_size,
                              hipStream_t stream) {
  (void)in_sizes; (void)n_in; (void)out_size; (void)ws_size;
  const float* x     = (const float*)d_in[0];
  const float* w_qkv = (const float*)d_in[1];
  const float* w_out = (const float*)d_in[2];
  const float* b_out = (const float*)d_in[3];
  float* out = (float*)d_out;

  char* ws = (char*)d_ws;
  size_t off = 0;
  auto alloc_h = [&](size_t halves) -> _Float16* {
    _Float16* p = (_Float16*)(ws + off);
    off += ((halves * 2 + 255) / 256) * 256;
    return p;
  };
  _Float16* xh  = alloc_h((size_t)ROWS * HID);
  _Float16* wqh = alloc_h((size_t)3 * HID * HID);
  _Float16* woh = alloc_h((size_t)HID * HID);
  _Float16* qh  = alloc_h((size_t)BATCH * NH * SEQ * HD);
  _Float16* kh  = alloc_h((size_t)BATCH * NH * SEQ * HD);
  _Float16* vth = alloc_h((size_t)BATCH * NH * SEQ * HD);
  _Float16* yh  = alloc_h((size_t)ROWS * HID);

  cast_f32_f16<<<2048, 256, 0, stream>>>(x, xh, ROWS * HID);
  cast_f32_f16<<<1024, 256, 0, stream>>>(w_qkv, wqh, 3 * HID * HID);
  cast_f32_f16<<<512, 256, 0, stream>>>(w_out, woh, HID * HID);

  // 4096/64 * 2304/64 = 2304 wave tiles / 4 waves per block
  qkv_gemm<<<576, 128, 0, stream>>>(xh, wqh, qh, kh, vth);
  // 24 (b*h) * 16 row-groups
  attn<<<384, 128, 0, stream>>>(qh, kh, vth, yh);
  // 4096/64 * 768/64 = 768 wave tiles / 4
  out_proj<<<192, 128, 0, stream>>>(yh, woh, b_out, out);
}
